// AttentionBlock_11879879544421
// MI455X (gfx1250) — matrix operand order repaired and checked by reading, not compiled, not run on hardware
//
#include <hip/hip_runtime.h>
#include <math.h>

typedef _Float16 f16;
typedef __attribute__((ext_vector_type(16))) _Float16 v16h;
typedef __attribute__((ext_vector_type(8)))  float    v8f;
typedef unsigned u32;
typedef __attribute__((ext_vector_type(4))) u32 u32x4;
typedef __attribute__((ext_vector_type(8))) u32 u32x8;

#define T_LEN   4096
#define C_CH    512
#define BATCH   4
#define QKV_C   1536            // 3*C
#define KQKV    32              // qkv conv kernel width = C/16
#define XPAD    (T_LEN + 32)    // left pad 15, right pad 17 (>=16)
#define HPAD    (T_LEN + 8)     // left pad 2, right pad 6 (taps padded 5->8)
#define HEADS   4
#define HCH     128             // channels per head

// ---------------------------------------------------------------------------
// Tensor Data Mover: 2-D f16 tile (tile_w x tile_h, row stride stride_elems)
// global -> LDS, async, tracked by TENSORcnt.  D# per ISA ch.8:
//  group0: count=1 | lds_addr | global byte addr (57b) | type=2
//  group1: data_size=1 (2B), tensor dims 1<<20 (no OOB), tile dims, stride.
// ---------------------------------------------------------------------------
__device__ __forceinline__ void tdm_load_tile_f16(
    unsigned lds_off, const void* gaddr, unsigned tile_w, unsigned tile_h,
    unsigned long long stride_elems) {
  unsigned long long ga = (unsigned long long)gaddr;
  u32x4 g0;
  g0.x = 1u;                                            // count=1, user mode
  g0.y = lds_off;                                       // lds_addr (bytes)
  g0.z = (u32)ga;                                       // global_addr[31:0]
  g0.w = (u32)((ga >> 32) & 0x01FFFFFFull) | (2u << 30);// addr[56:32]|type=2
  u32x8 g1;
  g1[0] = 1u << 16;                                     // data_size=2B
  g1[1] = 0u;                                           // tensor_dim0 lo16 (1<<20)
  g1[2] = 16u;                                          // dim0 hi | dim1 lo16
  g1[3] = 16u | (tile_w << 16);                         // dim1 hi | tile_dim0
  g1[4] = tile_h & 0xFFFFu;                             // tile_dim1 (tile_dim2=0)
  g1[5] = (u32)stride_elems;                            // dim0_stride[31:0]
  g1[6] = (u32)((stride_elems >> 32) & 0xFFFFull);      // dim0_stride[47:32]
  g1[7] = 0u;
  asm volatile("tensor_load_to_lds %0, %1" :: "s"(g0), "s"(g1) : "memory");
}

// ---------------------------------------------------------------------------
// fp32 -> fp16 convert (grid stride)
// ---------------------------------------------------------------------------
__global__ void f32_to_f16_kernel(const float* __restrict__ src,
                                  f16* __restrict__ dst, int n) {
  for (int i = blockIdx.x * 256 + threadIdx.x; i < n; i += gridDim.x * 256)
    dst[i] = (f16)src[i];
}

// pack proj weights [co][ci][5] -> f16 [co][ci][8] (tail zeroed by memset)
__global__ void pack_projw_kernel(const float* __restrict__ src,
                                  f16* __restrict__ dst) {
  int n = C_CH * C_CH * 5;
  for (int i = blockIdx.x * 256 + threadIdx.x; i < n; i += gridDim.x * 256) {
    int dk = i % 5, rest = i / 5;               // rest = co*512 + ci
    dst[(size_t)rest * 8 + dk] = (f16)src[i];
  }
}

// ---------------------------------------------------------------------------
// GroupNorm: one block per (b, group); 16 ch x 4096 t reduction
// ---------------------------------------------------------------------------
__global__ __launch_bounds__(256) void groupnorm_kernel(
    const float* __restrict__ x, const float* __restrict__ gamma,
    const float* __restrict__ beta, f16* __restrict__ xnp) {
  __shared__ float r1[256], r2[256];
  int b = blockIdx.x >> 5, g = blockIdx.x & 31;
  const float* xb = x + ((size_t)(b * C_CH + g * 16)) * T_LEN;
  float s = 0.f, s2 = 0.f;
  for (int j = threadIdx.x; j < 16 * T_LEN; j += 256) {
    float v = xb[j]; s += v; s2 += v * v;
  }
  r1[threadIdx.x] = s; r2[threadIdx.x] = s2;
  __syncthreads();
  for (int off = 128; off > 0; off >>= 1) {
    if (threadIdx.x < off) {
      r1[threadIdx.x] += r1[threadIdx.x + off];
      r2[threadIdx.x] += r2[threadIdx.x + off];
    }
    __syncthreads();
  }
  const float inv_n = 1.f / (16.f * T_LEN);
  float mu  = r1[0] * inv_n;
  float var = r2[0] * inv_n - mu * mu;
  float rsig = rsqrtf(var + 1e-5f);
  for (int j = threadIdx.x; j < 16 * T_LEN; j += 256) {
    int c = j >> 12, t = j & (T_LEN - 1);
    int ch = g * 16 + c;
    float v = (xb[j] - mu) * rsig * gamma[ch] + beta[ch];
    xnp[((size_t)(b * C_CH + ch)) * XPAD + 15 + t] = (f16)v;
  }
}

// ---------------------------------------------------------------------------
// QKV conv (k=32) as implicit GEMM via v_wmma_f32_16x16x32_f16.
// Each wave: 16(co) x 64(t) tile, K-loop over ci (K=32 taps per wmma).
// ---------------------------------------------------------------------------
__global__ __launch_bounds__(256) void qkv_conv_kernel(
    const f16* __restrict__ xnp, const f16* __restrict__ w,
    const float* __restrict__ bias, f16* __restrict__ qkv) {
  union F { v16h v; unsigned u[8]; _Float16 h[16]; };
  int wave = (blockIdx.x << 3) + (threadIdx.x >> 5);
  int lane = threadIdx.x & 31;
  int hi = lane >> 4, lo = lane & 15;
  const int TT = T_LEN / 64;                 // 64 t-tiles
  int b  = wave / (96 * TT);
  int r  = wave % (96 * TT);
  int co0 = (r / TT) * 16;
  int t0  = (r % TT) * 64;

  v8f acc[4] = {};
  const f16* wrow = w + ((size_t)(co0 + lo)) * (C_CH * KQKV);
  const f16* xrow = xnp + ((size_t)b * C_CH) * XPAD;

  for (int ci = 0; ci < C_CH; ++ci) {
    const f16* xr = xrow + (size_t)ci * XPAD + t0;
    __builtin_prefetch(xr + XPAD, 0, 1);     // next ci line -> global_prefetch
    F a;
    const unsigned* wa = (const unsigned*)(wrow + ci * KQKV);
#pragma unroll
    for (int i = 0; i < 8; ++i) {
      int k0 = (hi << 4) + (i << 1);  // A 16x32 layout
      a.u[i] = wa[k0 >> 1];
    }
#pragma unroll
    for (int nt = 0; nt < 4; ++nt) {
      F bm;
      int n = nt * 16 + lo;
#pragma unroll
      for (int i = 0; i < 8; ++i) {
        int dk0 = (hi << 4) + (i << 1);      // B 32x16 layout: K pairs
        bm.h[2 * i]     = xr[n + dk0];       // pad(15) + (t+dk-15) = t+dk
        bm.h[2 * i + 1] = xr[n + dk0 + 1];
      }
      acc[nt] = __builtin_amdgcn_wmma_f32_16x16x32_f16(
          false, a.v, false, bm.v, (short)0, acc[nt], false, false);
    }
  }
#pragma unroll
  for (int nt = 0; nt < 4; ++nt)
#pragma unroll
    for (int i = 0; i < 8; ++i) {
      int co = co0 + i + hi * 8;
      int t  = t0 + nt * 16 + lo;
      qkv[((size_t)b * QKV_C + co) * T_LEN + t] = (f16)(acc[nt][i] + bias[co]);
    }
}

// ---------------------------------------------------------------------------
// Flash attention: one wave per (head-batch, 16-query tile); s-chunks of 32.
// K tiles (128ch x 32s) are staged global->LDS by the Tensor Data Mover,
// double-buffered and overlapped with compute via TENSORcnt.
// ---------------------------------------------------------------------------
__global__ __launch_bounds__(32) void attention_kernel(
    const f16* __restrict__ qkv, f16* __restrict__ hpad) {
  union F { v16h v; unsigned u[8]; _Float16 h[16]; };
  __shared__ _Float16 skt[2][HCH * 32];      // K tiles [c][s], 8KB each
  __shared__ _Float16 pls[16 * 32];          // P tile for A-frag reshape
  int bh = blockIdx.x >> 8;                  // 16 head-batches
  int t0 = (blockIdx.x & 255) << 4;          // 256 query tiles of 16
  int b = bh >> 2, head = bh & 3;
  int lane = threadIdx.x, hi = lane >> 4, lo = lane & 15;
  const size_t base = ((size_t)b * QKV_C + head * 3 * HCH) * T_LEN;
  const f16* qp = qkv + base;
  const f16* kp = qp + (size_t)HCH * T_LEN;
  const f16* vp = qp + (size_t)(2 * HCH) * T_LEN;
  const float rs = 0.08838834764831845f;     // 1/sqrt(128) = scale^2

  // preload Q A-fragments (4 K-chunks of 32 channels), transposed gather
  F qa[4];
#pragma unroll
  for (int kc = 0; kc < 4; ++kc)
#pragma unroll
    for (int i = 0; i < 8; ++i) {
      int k0 = (hi << 4) + (i << 1);
      int c = kc * 32 + k0;
      qa[kc].h[2 * i]     = qp[(size_t)c * T_LEN + t0 + lo];
      qa[kc].h[2 * i + 1] = qp[(size_t)(c + 1) * T_LEN + t0 + lo];
    }

  v8f o[8] = {};
  float mrow[8], lrow[8];
#pragma unroll
  for (int i = 0; i < 8; ++i) { mrow[i] = -1e30f; lrow[i] = 0.f; }

  // TDM prologue: stage first K tile
  tdm_load_tile_f16((u32)(unsigned long long)(void*)&skt[0][0], kp,
                    32, HCH, (unsigned long long)T_LEN);

  int bi = 0;
  for (int s0 = 0; s0 < T_LEN; s0 += 32, bi ^= 1) {
    if (s0 + 32 < T_LEN) {                   // stage next tile, wait current
      tdm_load_tile_f16((u32)(unsigned long long)(void*)&skt[bi ^ 1][0],
                        kp + (s0 + 32), 32, HCH, (unsigned long long)T_LEN);
      __builtin_amdgcn_s_wait_tensorcnt(1);
    } else {
      __builtin_amdgcn_s_wait_tensorcnt(0);
    }
    asm volatile("" ::: "memory");
    const _Float16* sk = &skt[bi][0];

    v8f sc0 = {}, sc1 = {};
#pragma unroll
    for (int kc = 0; kc < 4; ++kc) {
      F b0, b1;
#pragma unroll
      for (int i = 0; i < 8; ++i) {
        int c = kc * 32 + (hi << 4) + (i << 1);
        b0.h[2 * i]     = sk[c * 32 + lo];
        b0.h[2 * i + 1] = sk[(c + 1) * 32 + lo];
        b1.h[2 * i]     = sk[c * 32 + 16 + lo];
        b1.h[2 * i + 1] = sk[(c + 1) * 32 + 16 + lo];
      }
      sc0 = __builtin_amdgcn_wmma_f32_16x16x32_f16(
          false, qa[kc].v, false, b0.v, (short)0, sc0, false, false);
      sc1 = __builtin_amdgcn_wmma_f32_16x16x32_f16(
          false, qa[kc].v, false, b1.v, (short)0, sc1, false, false);
    }
    // online softmax per row (row i+hi*8 lives on one 16-lane half)
#pragma unroll
    for (int i = 0; i < 8; ++i) {
      float v0 = sc0[i] * rs, v1 = sc1[i] * rs;
      float mx = fmaxf(v0, v1);
      for (int off = 1; off < 16; off <<= 1)
        mx = fmaxf(mx, __shfl_xor(mx, off, 32));
      float mnew = fmaxf(mrow[i], mx);
      float corr = __expf(mrow[i] - mnew);
      float p0 = __expf(v0 - mnew), p1 = __expf(v1 - mnew);
      float ps = p0 + p1;
      for (int off = 1; off < 16; off <<= 1)
        ps += __shfl_xor(ps, off, 32);
      lrow[i] = lrow[i] * corr + ps;
      mrow[i] = mnew;
#pragma unroll
      for (int ct = 0; ct < 8; ++ct) o[ct][i] *= corr;
      int rr = i + hi * 8;
      pls[rr * 32 + lo]      = (_Float16)p0;
      pls[rr * 32 + 16 + lo] = (_Float16)p1;
    }
    __syncthreads();
    F pa;                                    // P as A 16x32 fragment
#pragma unroll
    for (int i = 0; i < 8; ++i) {
      int k0 = (hi << 4) + (i << 1);
      pa.h[2 * i]     = pls[lo * 32 + k0];
      pa.h[2 * i + 1] = pls[lo * 32 + k0 + 1];
    }
#pragma unroll
    for (int ct = 0; ct < 8; ++ct) {         // P (16x32) x V^T (32x16)
      F vb;
      const f16* vcol = vp + (size_t)(ct * 16 + lo) * T_LEN + s0;
#pragma unroll
      for (int i = 0; i < 8; ++i)
        vb.u[i] = *(const unsigned*)(vcol + (hi << 4) + (i << 1));
      o[ct] = __builtin_amdgcn_wmma_f32_16x16x32_f16(
          false, pa.v, false, vb.v, (short)0, o[ct], false, false);
    }
    __syncthreads();
  }
#pragma unroll
  for (int i = 0; i < 8; ++i) {
    float inv = 1.f / lrow[i];
    int t = t0 + i + hi * 8;
#pragma unroll
    for (int ct = 0; ct < 8; ++ct) {
      int c = head * HCH + ct * 16 + lo;
      hpad[((size_t)b * C_CH + c) * HPAD + 2 + t] = (f16)(o[ct][i] * inv);
    }
  }
}

// ---------------------------------------------------------------------------
// Proj conv (k=5 padded to 8 taps) + bias + residual, WMMA K=32 = 4ci x 8taps
// ---------------------------------------------------------------------------
__global__ __launch_bounds__(256) void proj_conv_kernel(
    const f16* __restrict__ hpad, const f16* __restrict__ w8,
    const float* __restrict__ bias, const float* __restrict__ x,
    float* __restrict__ out) {
  union F { v16h v; unsigned u[8]; _Float16 h[16]; };
  int wave = (blockIdx.x << 3) + (threadIdx.x >> 5);
  int lane = threadIdx.x & 31;
  int hi = lane >> 4, lo = lane & 15;
  const int TT = T_LEN / 64;
  int b = wave / (32 * TT);
  int r = wave % (32 * TT);
  int co0 = (r / TT) * 16;
  int t0  = (r % TT) * 64;

  v8f acc[4] = {};
  const f16* wrow = w8 + (size_t)(co0 + lo) * (C_CH * 8);
  const f16* hrow = hpad + (size_t)b * C_CH * HPAD;

  for (int ci0 = 0; ci0 < C_CH; ci0 += 4) {
    F a;
    const unsigned* wa = (const unsigned*)(wrow + ci0 * 8);  // [ci][8] linear
#pragma unroll
    for (int i = 0; i < 8; ++i) {
      int k0 = (hi << 4) + (i << 1);
      a.u[i] = wa[k0 >> 1];
    }
#pragma unroll
    for (int nt = 0; nt < 4; ++nt) {
      F bm;
      int n = t0 + nt * 16 + lo;
#pragma unroll
      for (int i = 0; i < 8; ++i) {
        int kk = (hi << 4) + (i << 1);       // k = cig*8 + dk
        const f16* hp = hrow + (size_t)(ci0 + (kk >> 3)) * HPAD + n + (kk & 7);
        bm.h[2 * i] = hp[0];                 // pad(2)+(t+dk-2) = t+dk
        bm.h[2 * i + 1] = hp[1];
      }
      acc[nt] = __builtin_amdgcn_wmma_f32_16x16x32_f16(
          false, a.v, false, bm.v, (short)0, acc[nt], false, false);
    }
  }
#pragma unroll
  for (int nt = 0; nt < 4; ++nt)
#pragma unroll
    for (int i = 0; i < 8; ++i) {
      int co = co0 + i + hi * 8;
      int t  = t0 + nt * 16 + lo;
      size_t idx = ((size_t)b * C_CH + co) * T_LEN + t;
      out[idx] = x[idx] + bias[co] + acc[nt][i];
    }
}

// ---------------------------------------------------------------------------
extern "C" void kernel_launch(void* const* d_in, const int* in_sizes, int n_in,
                              void* d_out, int out_size, void* d_ws, size_t ws_size,
                              hipStream_t stream) {
  const float* x      = (const float*)d_in[0];
  const float* gamma  = (const float*)d_in[1];
  const float* beta   = (const float*)d_in[2];
  const float* qkv_w  = (const float*)d_in[3];
  const float* qkv_b  = (const float*)d_in[4];
  const float* proj_w = (const float*)d_in[5];
  const float* proj_b = (const float*)d_in[6];
  float* out = (float*)d_out;

  char* ws = (char*)d_ws;
  size_t off = 0;
  f16* xnp = (f16*)(ws + off); off += (size_t)BATCH * C_CH * XPAD * 2;
  f16* wq  = (f16*)(ws + off); off += (size_t)QKV_C * C_CH * KQKV * 2;
  f16* qkv = (f16*)(ws + off); off += (size_t)BATCH * QKV_C * T_LEN * 2;
  f16* hp  = (f16*)(ws + off); off += (size_t)BATCH * C_CH * HPAD * 2;
  f16* wp  = (f16*)(ws + off); off += (size_t)C_CH * C_CH * 8 * 2;

  hipMemsetAsync(xnp, 0, (size_t)BATCH * C_CH * XPAD * 2, stream);
  hipMemsetAsync(hp,  0, (size_t)BATCH * C_CH * HPAD * 2, stream);
  hipMemsetAsync(wp,  0, (size_t)C_CH * C_CH * 8 * 2, stream);

  f32_to_f16_kernel<<<4096, 256, 0, stream>>>(qkv_w, wq, QKV_C * C_CH * KQKV);
  pack_projw_kernel<<<2048, 256, 0, stream>>>(proj_w, wp);
  groupnorm_kernel<<<BATCH * 32, 256, 0, stream>>>(x, gamma, beta, xnp);
  qkv_conv_kernel<<<3072, 256, 0, stream>>>(xnp, wq, qkv_b, qkv);     // 24576 waves
  attention_kernel<<<16 * 256, 32, 0, stream>>>(qkv, hp);             // 4096 waves
  proj_conv_kernel<<<1024, 256, 0, stream>>>(hp, wp, proj_b, x, out); // 8192 waves
}
